// DLRM_81767587381517
// MI455X (gfx1250) — matrix operand order repaired and checked by reading, not compiled, not run on hardware
//
#include <hip/hip_runtime.h>
#include <math.h>

// ---------------------------------------------------------------------------
// DLRM-style recommender forward for MI455X (gfx1250), wave32 + WMMA f16.
// ---------------------------------------------------------------------------

#define DEV static __device__ __forceinline__

typedef __attribute__((ext_vector_type(16))) _Float16 v16h;
typedef __attribute__((ext_vector_type(8)))  float    v8f;

#define D_        28
#define L_        50
#define PAD_IDX   59047
#define UPAD_IDX  162541
#define RSCALE    0.1889822365046136f   // 1/sqrt(28)

DEV int lane_id() { return (int)(threadIdx.x & 31u); }

DEV v8f vzero8() {
  v8f v;
#pragma unroll
  for (int i = 0; i < 8; ++i) v[i] = 0.0f;
  return v;
}

// D = A(16x32) * B(32x16) + C  via v_wmma_f32_16x16x32_f16
DEV v8f wmma16(v16h a, v16h b, v8f c) {
  return __builtin_amdgcn_wmma_f32_16x16x32_f16(false, a, false, b,
                                                (short)0, c, false, false);
}

// --- A fragment: 16x32 f16 tile, row-major in LDS (leading dim ld halfs) ---
// ISA layout: lanes 0-15 hold row M=lane, K={0..7,16..23}; lanes 16-31 hold
// row M=lane-16, K={8..15,24..31}.
DEV v16h frag_a_h(const _Float16* A, int ld) {
  int l = lane_id(), r = l & 15, hh = l >> 4;
  const _Float16* p = A + r * ld + 8 * hh;
  v16h a;
#pragma unroll
  for (int i = 0; i < 8; ++i) a[i] = p[i];
#pragma unroll
  for (int i = 0; i < 8; ++i) a[8 + i] = p[16 + i];
  return a;
}

DEV v16h frag_a_f32(const float* A, int ld) {
  int l = lane_id(), r = l & 15, hh = l >> 4;
  const float* p = A + r * ld + 8 * hh;
  v16h a;
#pragma unroll
  for (int i = 0; i < 8; ++i) a[i] = (_Float16)p[i];
#pragma unroll
  for (int i = 0; i < 8; ++i) a[8 + i] = (_Float16)p[16 + i];
  return a;
}

// --- global f32 A fragments (full 16x32 tile in bounds) ---
// scalar-load variant (any alignment)
DEV v16h frag_a_gu(const float* A, int ld) {
  int l = lane_id(), r = l & 15, hh = l >> 4;
  const float* p = A + r * ld + 8 * hh;
  v16h a;
#pragma unroll
  for (int i = 0; i < 8; ++i) a[i] = (_Float16)p[i];
#pragma unroll
  for (int i = 0; i < 8; ++i) a[8 + i] = (_Float16)p[16 + i];
  return a;
}

// float4 variant: requires (tile base + r*ld) to be 16B aligned, i.e. the
// buffer is 16B aligned and ld % 4 == 0 (checked wave-uniformly by caller).
// Emits 4x global_load_b128 per fragment instead of 16x b32.
DEV v16h frag_a_gu4(const float* A, int ld) {
  int l = lane_id(), r = l & 15, hh = l >> 4;
  const float* base = A + r * ld + 8 * hh;
  float4 x0 = *(const float4*)(base + 0);
  float4 x1 = *(const float4*)(base + 4);
  float4 x2 = *(const float4*)(base + 16);
  float4 x3 = *(const float4*)(base + 20);
  v16h a;
  a[0] = (_Float16)x0.x;  a[1] = (_Float16)x0.y;
  a[2] = (_Float16)x0.z;  a[3] = (_Float16)x0.w;
  a[4] = (_Float16)x1.x;  a[5] = (_Float16)x1.y;
  a[6] = (_Float16)x1.z;  a[7] = (_Float16)x1.w;
  a[8] = (_Float16)x2.x;  a[9] = (_Float16)x2.y;
  a[10] = (_Float16)x2.z; a[11] = (_Float16)x2.w;
  a[12] = (_Float16)x3.x; a[13] = (_Float16)x3.y;
  a[14] = (_Float16)x3.z; a[15] = (_Float16)x3.w;
  return a;
}

// edge A fragment: clamped addresses (always-legal unconditional loads) +
// value selects; runs only on ragged tiles.
DEV v16h frag_a_edge(const float* A, int ld, int rmax, int kmax) {
  int l = lane_id(), r = l & 15, hh = l >> 4;
  bool rok = r < rmax;
  int rbase = rok ? r * ld : 0;
  v16h a;
#pragma unroll
  for (int i = 0; i < 8; ++i) {
    int k = 8 * hh + i;
    float x = A[rbase + ((k < kmax) ? k : 0)];
    a[i] = (_Float16)((rok && k < kmax) ? x : 0.f);
  }
#pragma unroll
  for (int i = 0; i < 8; ++i) {
    int k = 16 + 8 * hh + i;
    float x = A[rbase + ((k < kmax) ? k : 0)];
    a[8 + i] = (_Float16)((rok && k < kmax) ? x : 0.f);
  }
  return a;
}

// --- B fragment from weight layout W[N][K] row-major (== B^T), f16 LDS ----
// ISA layout: lanes 0-15 hold K={0..7,16..23}; lanes 16-31 hold K={8..15,24..31},
// lane%16 = column n.
DEV v16h frag_bW_h(const _Float16* W, int ld) {
  int l = lane_id(), n = l & 15, hh = l >> 4;
  const _Float16* p = W + n * ld + 8 * hh;
  v16h b;
#pragma unroll
  for (int j = 0; j < 16; ++j) b[j] = p[j < 8 ? j : j + 8];
  return b;
}

// global f32 B fragments (full tile in bounds)
DEV v16h frag_bW_gu(const float* W, int ld) {
  int l = lane_id(), n = l & 15, hh = l >> 4;
  const float* p = W + n * ld + 8 * hh;
  v16h b;
#pragma unroll
  for (int j = 0; j < 16; ++j) b[j] = (_Float16)p[j < 8 ? j : j + 8];
  return b;
}

DEV v16h frag_bW_gu4(const float* W, int ld) {
  int l = lane_id(), n = l & 15, hh = l >> 4;
  const float* base = W + n * ld + 8 * hh;
  float4 x0 = *(const float4*)(base + 0);
  float4 x1 = *(const float4*)(base + 4);
  float4 x2 = *(const float4*)(base + 16);
  float4 x3 = *(const float4*)(base + 20);
  v16h b;
  b[0] = (_Float16)x0.x;  b[1] = (_Float16)x0.y;
  b[2] = (_Float16)x0.z;  b[3] = (_Float16)x0.w;
  b[4] = (_Float16)x1.x;  b[5] = (_Float16)x1.y;
  b[6] = (_Float16)x1.z;  b[7] = (_Float16)x1.w;
  b[8] = (_Float16)x2.x;  b[9] = (_Float16)x2.y;
  b[10] = (_Float16)x2.z; b[11] = (_Float16)x2.w;
  b[12] = (_Float16)x3.x; b[13] = (_Float16)x3.y;
  b[14] = (_Float16)x3.z; b[15] = (_Float16)x3.w;
  return b;
}

DEV v16h frag_bW_edge(const float* W, int ld, int nmax, int kmax) {
  int l = lane_id(), n = l & 15, hh = l >> 4;
  bool nok = n < nmax;
  int nbase = nok ? n * ld : 0;
  v16h b;
#pragma unroll
  for (int j = 0; j < 16; ++j) {
    int k = 8 * hh + (j < 8 ? j : j + 8);
    float x = W[nbase + ((k < kmax) ? k : 0)];
    b[j] = (_Float16)((nok && k < kmax) ? x : 0.f);
  }
  return b;
}

// ===========================================================================
// Generic GEMM: Y[M,N] = act(X[M,K] @ W[N,K]^T + bias), fp32 I/O, f16 WMMA.
// grid.x tiles M by 128 (8 waves x 16 rows), grid.y tiles N by 128.
// ===========================================================================
__global__ void k_gemm(const float* __restrict__ X, const float* __restrict__ W,
                       const float* __restrict__ bias, float* __restrict__ Y,
                       int M, int N, int K, int act) {
  int wave = (int)(threadIdx.x >> 5);
  int lane = lane_id();
  int m0 = (blockIdx.x * 8 + wave) * 16;
  if (m0 >= M) return;
  int n0base = blockIdx.y * 128;
  int rmax = M - m0;
  bool v4 = ((K & 3) == 0);     // wave-uniform: rows 16B-aligned -> b128 loads

  v8f acc[8];
#pragma unroll
  for (int j = 0; j < 8; ++j) acc[j] = vzero8();

  const float* Arow = X + (size_t)m0 * K;
  for (int k0 = 0; k0 < K; k0 += 32) {
    if (k0 + 32 < K) __builtin_prefetch(Arow + k0 + 32, 0, 1);
    int kmax = K - k0;
    v16h a;
    if (rmax >= 16 && kmax >= 32)
      a = v4 ? frag_a_gu4(Arow + k0, K) : frag_a_gu(Arow + k0, K);
    else
      a = frag_a_edge(Arow + k0, K, rmax, kmax);
#pragma unroll
    for (int j = 0; j < 8; ++j) {
      int n0 = n0base + j * 16;
      if (n0 < N) {
        const float* Wt = W + (size_t)n0 * K + k0;
        int nmax = N - n0;
        v16h b;
        if (nmax >= 16 && kmax >= 32)
          b = v4 ? frag_bW_gu4(Wt, K) : frag_bW_gu(Wt, K);
        else
          b = frag_bW_edge(Wt, K, nmax, kmax);
        acc[j] = wmma16(a, b, acc[j]);
      }
    }
  }

  int n = lane & 15, hh = lane >> 4;
#pragma unroll
  for (int j = 0; j < 8; ++j) {
    int nn = n0base + j * 16 + n;
    if (nn >= N) continue;
    float bv = bias ? bias[nn] : 0.f;
#pragma unroll
    for (int r = 0; r < 8; ++r) {
      int m = m0 + r + 8 * hh;
      if (m >= M) continue;
      float v = acc[j][r] + bv;
      if (act) v = fmaxf(v, 0.f);
      Y[(size_t)m * N + nn] = v;
    }
  }
}

// ===========================================================================
// Per-sample causal self-attention over the 50-item history. One block/sample,
// 128 threads (4 waves); wave w owns M-tile w of the padded 64-row matrices.
// ===========================================================================
__global__ void k_attn(const int* __restrict__ history,
                       const float* __restrict__ he,
                       const float* __restrict__ qW, const float* __restrict__ kW,
                       const float* __restrict__ vW,
                       float* __restrict__ hist_e /* [B,50,28] */) {
  __shared__ _Float16 rawh[64 * 32];
  __shared__ _Float16 wq[32 * 32], wk[32 * 32], wv[32 * 32];
  __shared__ _Float16 Qh[64 * 32], Kh[64 * 32], VT[32 * 64];
  __shared__ float    S[64 * 64];
  __shared__ int      hidx[64];

  int b = blockIdx.x;
  int tid = (int)threadIdx.x;
  int wave = tid >> 5, lane = lane_id();

  for (int i = tid; i < 32 * 32; i += 128) {
    int n = i >> 5, k = i & 31;
    float q = 0.f, kk = 0.f, v = 0.f;
    if (n < D_ && k < D_) { q = qW[n * D_ + k]; kk = kW[n * D_ + k]; v = vW[n * D_ + k]; }
    wq[i] = (_Float16)q; wk[i] = (_Float16)kk; wv[i] = (_Float16)v;
  }
  if (tid < 64) hidx[tid] = (tid < L_) ? history[(size_t)b * L_ + tid] : PAD_IDX;
  __syncthreads();

  for (int i = tid; i < 64 * 32; i += 128) {
    int l = i >> 5, d = i & 31;
    float v = 0.f;
    if (l < L_ && d < D_) v = he[(size_t)hidx[l] * D_ + d];
    rawh[i] = (_Float16)v;
  }
  __syncthreads();

  // ---- Q, K, V = rawh @ {q,k,v}W^T  (K=28->32, one WMMA K-step) ----
  {
    v16h a = frag_a_h(rawh + wave * 16 * 32, 32);
#pragma unroll
    for (int nt = 0; nt < 2; ++nt) {
      v16h bq = frag_bW_h(wq + nt * 16 * 32, 32);
      v16h bk = frag_bW_h(wk + nt * 16 * 32, 32);
      v16h bv = frag_bW_h(wv + nt * 16 * 32, 32);
      v8f cq = wmma16(a, bq, vzero8());
      v8f ck = wmma16(a, bk, vzero8());
      v8f cv = wmma16(a, bv, vzero8());
      int n = lane & 15, hh = lane >> 4;
#pragma unroll
      for (int r = 0; r < 8; ++r) {
        int m = wave * 16 + r + 8 * hh;
        int nn = nt * 16 + n;
        Qh[m * 32 + nn] = (_Float16)cq[r];
        Kh[m * 32 + nn] = (_Float16)ck[r];
        VT[nn * 64 + m] = (_Float16)cv[r];   // store V transposed: [d][m]
      }
    }
  }
  __syncthreads();

  // ---- S = Q @ K^T * scale + masks.  K-matrix rows are exactly W-layout ----
  {
    v16h a = frag_a_h(Qh + wave * 16 * 32, 32);
#pragma unroll
    for (int nt = 0; nt < 4; ++nt) {
      v16h bb = frag_bW_h(Kh + nt * 16 * 32, 32);
      v8f c = wmma16(a, bb, vzero8());
      int n = lane & 15, hh = lane >> 4;
#pragma unroll
      for (int r = 0; r < 8; ++r) {
        int i = wave * 16 + r + 8 * hh;    // query row
        int j = nt * 16 + n;               // key col
        float s = c[r] * RSCALE;
        bool pad = (j >= L_) || (hidx[j] == PAD_IDX);
        if (pad) s = -10000.f;
        else if (j > i) s += -10000.f;     // causal mask
        S[i * 64 + j] = s;
      }
    }
  }
  __syncthreads();

  // ---- row softmax (64-wide, masked cols underflow to exactly 0) ----
  if (tid < 64) {
    float mx = -1e30f;
    for (int j = 0; j < 64; ++j) mx = fmaxf(mx, S[tid * 64 + j]);
    float sum = 0.f;
    for (int j = 0; j < 64; ++j) { float e = expf(S[tid * 64 + j] - mx); S[tid * 64 + j] = e; sum += e; }
    float inv = 1.f / sum;
    for (int j = 0; j < 64; ++j) S[tid * 64 + j] *= inv;
  }
  __syncthreads();

  // ---- hist_e = P @ V + raw  (K=50->64, two WMMA K-steps; B = V^T) ----
  {
    v8f acc[2] = { vzero8(), vzero8() };
#pragma unroll
    for (int kc = 0; kc < 2; ++kc) {
      v16h a = frag_a_f32(S + wave * 16 * 64 + kc * 32, 64);
#pragma unroll
      for (int nt = 0; nt < 2; ++nt) {
        v16h bb = frag_bW_h(VT + nt * 16 * 64 + kc * 32, 64);
        acc[nt] = wmma16(a, bb, acc[nt]);
      }
    }
    int n = lane & 15, hh = lane >> 4;
#pragma unroll
    for (int nt = 0; nt < 2; ++nt)
#pragma unroll
      for (int r = 0; r < 8; ++r) {
        int m = wave * 16 + r + 8 * hh;
        int d = nt * 16 + n;
        if (m < L_ && d < D_)
          hist_e[((size_t)b * L_ + m) * D_ + d] = acc[nt][r] + (float)rawh[m * 32 + d];
      }
  }
}

// ===========================================================================
// Fused DIN attention: builds X=[h|t|h*t] (50x168 -> 64x192 f16) in LDS,
// runs [64,192]x[192,64] WMMA GEMM + relu, 64->1 projection, masked softmax,
// weighted sum of the 28-dim history vectors. One block (4 waves) per sample.
// IS_USER=1: h from precomputed hist_e, query=item_e, pad=PAD_IDX.
// IS_USER=0: h gathered from rater table, query=user_e, pad=UPAD_IDX.
// ===========================================================================
template <int IS_USER>
__global__ void k_din(const float* __restrict__ hsrc,  // [B,50,28] (user side)
                      const float* __restrict__ emb,   // rater table (item side)
                      const int*   __restrict__ idx,   // [B,50]
                      const float* __restrict__ ratings,
                      const float* __restrict__ query, // [B,28]
                      const float* __restrict__ rpW, const float* __restrict__ rpb,
                      const float* __restrict__ W1, const float* __restrict__ b1,
                      const float* __restrict__ W2, const float* __restrict__ b2,
                      int padIdx,
                      float* __restrict__ outv /* [B,28] */) {
  __shared__ _Float16 X[64 * 192];
  __shared__ _Float16 W1l[64 * 192];
  __shared__ _Float16 H[64 * 64];
  __shared__ float    qv[D_];
  __shared__ float    wrow[64];

  int b = blockIdx.x;
  int tid = (int)threadIdx.x;
  int wave = tid >> 5, lane = lane_id();

  if (tid < D_) qv[tid] = query[(size_t)b * D_ + tid];

  // h-part: cols [0,28)=embed, [28,56)=rating projection
  for (int i = tid; i < 64 * 56; i += 128) {
    int l = i / 56, c = i % 56;
    float v = 0.f;
    if (l < L_) {
      if (c < D_) {
        if (IS_USER) v = hsrc[((size_t)b * L_ + l) * D_ + c];
        else         v = emb[(size_t)idx[(size_t)b * L_ + l] * D_ + c];
      } else {
        int d = c - D_;
        v = ratings[(size_t)b * L_ + l] * rpW[d] + rpb[d];
      }
    }
    X[l * 192 + c] = (_Float16)v;
  }
  for (int i = tid; i < 64 * 192; i += 128) {
    int n = i / 192, k = i % 192;
    W1l[i] = (_Float16)((k < 168) ? W1[n * 168 + k] : 0.f);
  }
  __syncthreads();

  // t-part and h*t-part
  for (int i = tid; i < 64 * 136; i += 128) {
    int l = i / 136, c = 56 + i % 136;
    float v;
    if (c < 84)       v = qv[c - 56];
    else if (c < 112) v = qv[c - 84];
    else if (c < 140) v = (float)X[l * 192 + (c - 112)] * qv[c - 112];
    else if (c < 168) v = (float)X[l * 192 + (c - 140 + D_)] * qv[c - 140];
    else              v = 0.f;
    X[l * 192 + c] = (_Float16)v;
  }
  __syncthreads();

  // [64,192] x [192,64] : 6 K-steps x 4 N-tiles per wave (24 WMMAs/wave)
  {
    v8f acc[4] = { vzero8(), vzero8(), vzero8(), vzero8() };
#pragma unroll
    for (int kc = 0; kc < 6; ++kc) {
      v16h a = frag_a_h(X + wave * 16 * 192 + kc * 32, 192);
#pragma unroll
      for (int nt = 0; nt < 4; ++nt) {
        v16h bb = frag_bW_h(W1l + nt * 16 * 192 + kc * 32, 192);
        acc[nt] = wmma16(a, bb, acc[nt]);
      }
    }
    int n = lane & 15, hh = lane >> 4;
#pragma unroll
    for (int nt = 0; nt < 4; ++nt)
#pragma unroll
      for (int r = 0; r < 8; ++r) {
        int m = wave * 16 + r + 8 * hh;
        int nn = nt * 16 + n;
        H[m * 64 + nn] = (_Float16)fmaxf(acc[nt][r] + b1[nn], 0.f);
      }
  }
  __syncthreads();

  if (tid < 64) {
    float s = b2[0];
    for (int j = 0; j < 64; ++j) s += (float)H[tid * 64 + j] * W2[j];
    bool pad = (tid >= L_) || (idx[(size_t)b * L_ + tid] == padIdx);
    wrow[tid] = pad ? -10000.f : s;
  }
  __syncthreads();

  if (tid < D_) {
    float mx = -1e30f;
    for (int l = 0; l < L_; ++l) mx = fmaxf(mx, wrow[l]);
    float sum = 0.f;
    for (int l = 0; l < L_; ++l) sum += expf(wrow[l] - mx);
    float inv = 1.f / sum;
    float o = 0.f;
    for (int l = 0; l < L_; ++l)
      o += expf(wrow[l] - mx) * inv * (float)X[l * 192 + tid];
    outv[(size_t)b * D_ + tid] = o;
  }
}

// ===========================================================================
// Small VALU kernels
// ===========================================================================
__global__ void k_gather(const int* __restrict__ ids, const float* __restrict__ tab,
                         float* __restrict__ out, long n) {
  long i = (long)blockIdx.x * blockDim.x + threadIdx.x;
  if (i >= n) return;
  long row = i / D_; int d = (int)(i % D_);
  out[i] = tab[(size_t)ids[row] * D_ + d];
}

__global__ void k_gate(const float* __restrict__ gl, const float* __restrict__ ge,
                       const float* __restrict__ ie, float* __restrict__ gf, long n) {
  long i = (long)blockIdx.x * blockDim.x + threadIdx.x;
  if (i >= n) return;
  float g = 1.f / (1.f + expf(-gl[i]));
  gf[i] = g * ge[i] + (1.f - g) * ie[i];
}

__global__ void k_concat3(const float* a, const float* b, const float* c,
                          float* o, int Bn) {
  long i = (long)blockIdx.x * blockDim.x + threadIdx.x;
  if (i >= (long)Bn * 84) return;
  long bb = i / 84; int cc = (int)(i % 84);
  float v = (cc < 28) ? a[bb * 28 + cc]
          : (cc < 56) ? b[bb * 28 + cc - 28]
                      : c[bb * 28 + cc - 56];
  o[i] = v;
}

__global__ void k_concat4(const float* a, const float* b, const float* c,
                          const float* d, float* o, int Bn) {
  long i = (long)blockIdx.x * blockDim.x + threadIdx.x;
  if (i >= (long)Bn * 112) return;
  long bb = i / 112; int cc = (int)(i % 112);
  float v = (cc < 28) ? a[bb * 28 + cc]
          : (cc < 56) ? b[bb * 28 + cc - 28]
          : (cc < 84) ? c[bb * 28 + cc - 56]
                      : d[bb * 28 + cc - 84];
  o[i] = v;
}

__global__ void k_fusetop(const float* __restrict__ us, const float* __restrict__ is_,
                          float* __restrict__ xtop, int Bn) {
  long i = (long)blockIdx.x * blockDim.x + threadIdx.x;
  if (i >= (long)Bn * 64) return;
  long b = i / 64; int j = (int)(i % 64);
  float u = us[b * 64 + j], v = is_[b * 64 + j];
  float* row = xtop + (size_t)b * 388;
  row[196 + j] = u; row[260 + j] = v; row[324 + j] = u * v;
}

__global__ void k_final(const float* __restrict__ t3, const float* __restrict__ W4,
                        const float* __restrict__ b4, float* __restrict__ y, int Bn) {
  int b = blockIdx.x * blockDim.x + threadIdx.x;
  if (b >= Bn) return;
  float s = b4[0];
  for (int k = 0; k < 64; ++k) s += t3[(size_t)b * 64 + k] * W4[k];
  y[b] = s;
}

DEV const float* pick7(int n, const float* a0, const float* a1, const float* a2,
                       const float* a3, const float* a4, const float* a5,
                       const float* a6) {
  switch (n) {
    case 0: return a0; case 1: return a1; case 2: return a2;
    case 3: return a3; case 4: return a4; case 5: return a5;
    default: return a6;
  }
}

// 7-field single-head attention (tiny, VALU). Block = 224 threads / sample.
__global__ void k_fieldattn(const float* ue_, const float* ie_, const float* uh,
                            const float* ihe, const float* de, const float* ge_,
                            const float* gf, const float* Win, const float* bin,
                            const float* Wout, const float* bout,
                            float* __restrict__ xtop /* [B,388] */) {
  __shared__ float F[7 * 28], FQ[7 * 28], FK[7 * 28], FV[7 * 28];
  __shared__ float P[7 * 7], T1[7 * 28];
  int b = blockIdx.x, tid = (int)threadIdx.x;

  if (tid < 196) {
    int n = tid / 28, d = tid % 28;
    F[tid] = pick7(n, ue_, ie_, uh, ihe, de, ge_, gf)[(size_t)b * 28 + d];
  }
  __syncthreads();
  if (tid < 196) {
    int n = tid / 28, d = tid % 28;
    float q = bin[d], k = bin[28 + d], v = bin[56 + d];
    for (int c = 0; c < 28; ++c) {
      float f = F[n * 28 + c];
      q += f * Win[d * 28 + c];
      k += f * Win[(28 + d) * 28 + c];
      v += f * Win[(56 + d) * 28 + c];
    }
    FQ[tid] = q; FK[tid] = k; FV[tid] = v;
  }
  __syncthreads();
  if (tid < 49) {
    int n = tid / 7, m = tid % 7;
    float s = 0.f;
    for (int c = 0; c < 28; ++c) s += FQ[n * 28 + c] * FK[m * 28 + c];
    P[tid] = s * RSCALE;
  }
  __syncthreads();
  if (tid < 7) {
    float mx = -1e30f;
    for (int m = 0; m < 7; ++m) mx = fmaxf(mx, P[tid * 7 + m]);
    float sum = 0.f;
    for (int m = 0; m < 7; ++m) { float e = expf(P[tid * 7 + m] - mx); P[tid * 7 + m] = e; sum += e; }
    float inv = 1.f / sum;
    for (int m = 0; m < 7; ++m) P[tid * 7 + m] *= inv;
  }
  __syncthreads();
  if (tid < 196) {
    int n = tid / 28, d = tid % 28;
    float s = 0.f;
    for (int m = 0; m < 7; ++m) s += P[n * 7 + m] * FV[m * 28 + d];
    T1[tid] = s;
  }
  __syncthreads();
  if (tid < 196) {
    int n = tid / 28, d = tid % 28;
    float o = bout[d];
    for (int c = 0; c < 28; ++c) o += T1[n * 28 + c] * Wout[d * 28 + c];
    xtop[(size_t)b * 388 + tid] = F[tid] + o;
  }
}

// ===========================================================================
// Host launcher
// ===========================================================================
extern "C" void kernel_launch(void* const* d_in, const int* in_sizes, int n_in,
                              void* d_out, int out_size, void* d_ws, size_t ws_size,
                              hipStream_t stream) {
  (void)n_in; (void)out_size; (void)ws_size;
  const int Bn = in_sizes[0];

  const int*   user_id   = (const int*)d_in[0];
  const int*   movie_id  = (const int*)d_in[1];
  const float* dense     = (const float*)d_in[2];
  const int*   history   = (const int*)d_in[3];
  const float* hist_rat  = (const float*)d_in[4];
  const float* genres    = (const float*)d_in[5];
  const int*   item_hist = (const int*)d_in[6];
  const float* ihist_rat = (const float*)d_in[7];
  const float* genome    = (const float*)d_in[8];
  // d_in[9] = has_genome_mask (unused, as in reference)
  const float* ue     = (const float*)d_in[10];
  const float* ie     = (const float*)d_in[11];
  const float* he     = (const float*)d_in[12];
  const float* re     = (const float*)d_in[13];
  const float* rp_W   = (const float*)d_in[14];
  const float* rp_b   = (const float*)d_in[15];
  const float* q_W    = (const float*)d_in[16];
  const float* k_W    = (const float*)d_in[17];
  const float* v_W    = (const float*)d_in[18];
  const float* din_W1 = (const float*)d_in[19];
  const float* din_b1 = (const float*)d_in[20];
  const float* din_W2 = (const float*)d_in[21];
  const float* din_b2 = (const float*)d_in[22];
  const float* idin_W1= (const float*)d_in[23];
  const float* idin_b1= (const float*)d_in[24];
  const float* idin_W2= (const float*)d_in[25];
  const float* idin_b2= (const float*)d_in[26];
  const float* bm_W1  = (const float*)d_in[27];
  const float* bm_b1  = (const float*)d_in[28];
  const float* bm_W2  = (const float*)d_in[29];
  const float* bm_b2  = (const float*)d_in[30];
  const float* gp_W   = (const float*)d_in[31];
  const float* gp_b   = (const float*)d_in[32];
  const float* gn_W1  = (const float*)d_in[33];
  const float* gn_b1  = (const float*)d_in[34];
  const float* gn_W2  = (const float*)d_in[35];
  const float* gn_b2  = (const float*)d_in[36];
  const float* gn_W3  = (const float*)d_in[37];
  const float* gn_b3  = (const float*)d_in[38];
  const float* gg_W   = (const float*)d_in[39];
  const float* gg_b   = (const float*)d_in[40];
  const float* fa_in_W  = (const float*)d_in[41];
  const float* fa_in_b  = (const float*)d_in[42];
  const float* fa_out_W = (const float*)d_in[43];
  const float* fa_out_b = (const float*)d_in[44];
  const float* us_W1  = (const float*)d_in[45];
  const float* us_b1  = (const float*)d_in[46];
  const float* us_W2  = (const float*)d_in[47];
  const float* us_b2  = (const float*)d_in[48];
  const float* is_W1  = (const float*)d_in[49];
  const float* is_b1  = (const float*)d_in[50];
  const float* is_W2  = (const float*)d_in[51];
  const float* is_b2  = (const float*)d_in[52];
  const float* tm_W1  = (const float*)d_in[53];
  const float* tm_b1  = (const float*)d_in[54];
  const float* tm_W2  = (const float*)d_in[55];
  const float* tm_b2  = (const float*)d_in[56];
  const float* tm_W3  = (const float*)d_in[57];
  const float* tm_b3  = (const float*)d_in[58];
  const float* tm_W4  = (const float*)d_in[59];
  const float* tm_b4  = (const float*)d_in[60];

  // workspace carve-up (floats); big scratch buffers are reused sequentially
  float* w = (float*)d_ws;
  size_t off = 0;
  auto alloc = [&](size_t n) { float* p = w + off; off += n; return p; };
  float* hist_e  = alloc((size_t)Bn * L_ * D_);
  float* user_e  = alloc((size_t)Bn * D_);
  float* item_e  = alloc((size_t)Bn * D_);
  float* user_h  = alloc((size_t)Bn * D_);
  float* item_he = alloc((size_t)Bn * D_);
  float* dense_e = alloc((size_t)Bn * D_);
  float* genre_e = alloc((size_t)Bn * D_);
  float* ge      = alloc((size_t)Bn * D_);
  float* gl      = alloc((size_t)Bn * D_);
  float* gf      = alloc((size_t)Bn * D_);
  float* g2      = alloc((size_t)Bn * 64);
  float* us      = alloc((size_t)Bn * 64);
  float* is_     = alloc((size_t)Bn * 64);
  float* t3      = alloc((size_t)Bn * 64);
  float* us_in   = alloc((size_t)Bn * 84);
  float* is_in   = alloc((size_t)Bn * 112);
  float* xtop    = alloc((size_t)Bn * 388);
  float* s256    = alloc((size_t)Bn * 256);  // g1 / us_h / is_h / t1
  float* s128    = alloc((size_t)Bn * 128);  // dense_h / t2

  auto gemm = [&](const float* Xp, const float* Wp, const float* bp, float* Yp,
                  int M, int N, int K, int act) {
    dim3 g((unsigned)((M + 127) / 128), (unsigned)((N + 127) / 128));
    k_gemm<<<g, 256, 0, stream>>>(Xp, Wp, bp, Yp, M, N, K, act);
  };
  auto egrid = [](long n) { return dim3((unsigned)((n + 255) / 256)); };

  // 0) id-embedding gathers
  k_gather<<<egrid((long)Bn * D_), 256, 0, stream>>>(user_id, ue, user_e, (long)Bn * D_);
  k_gather<<<egrid((long)Bn * D_), 256, 0, stream>>>(movie_id, ie, item_e, (long)Bn * D_);

  // 1) causal self-attention over history -> hist_e
  k_attn<<<Bn, 128, 0, stream>>>(history, he, q_W, k_W, v_W, hist_e);

  // 2) fused DIN attentions (user side & item side)
  k_din<1><<<Bn, 128, 0, stream>>>(hist_e, nullptr, history, hist_rat, item_e,
                                   rp_W, rp_b, din_W1, din_b1, din_W2, din_b2,
                                   PAD_IDX, user_h);
  k_din<0><<<Bn, 128, 0, stream>>>(nullptr, re, item_hist, ihist_rat, user_e,
                                   rp_W, rp_b, idin_W1, idin_b1, idin_W2, idin_b2,
                                   UPAD_IDX, item_he);

  // 3) dense / genre / genome MLPs (WMMA GEMMs)
  gemm(dense, bm_W1, bm_b1, s128, Bn, 128, 17, 1);
  gemm(s128, bm_W2, bm_b2, dense_e, Bn, 28, 128, 1);
  gemm(genres, gp_W, gp_b, genre_e, Bn, 28, 20, 1);
  gemm(genome, gn_W1, gn_b1, s256, Bn, 256, 1128, 1);
  gemm(s256, gn_W2, gn_b2, g2, Bn, 64, 256, 1);
  gemm(g2, gn_W3, gn_b3, ge, Bn, 28, 64, 0);
  gemm(ge, gg_W, gg_b, gl, Bn, 28, 28, 0);
  k_gate<<<egrid((long)Bn * D_), 256, 0, stream>>>(gl, ge, item_e, gf, (long)Bn * D_);

  // 4) field attention -> xtop[:, 0:196]
  k_fieldattn<<<Bn, 224, 0, stream>>>(user_e, item_e, user_h, item_he, dense_e,
                                      genre_e, gf, fa_in_W, fa_in_b, fa_out_W,
                                      fa_out_b, xtop);

  // 5) user/item side towers
  k_concat3<<<egrid((long)Bn * 84), 256, 0, stream>>>(user_e, user_h, dense_e, us_in, Bn);
  gemm(us_in, us_W1, us_b1, s256, Bn, 256, 84, 1);
  gemm(s256, us_W2, us_b2, us, Bn, 64, 256, 1);
  k_concat4<<<egrid((long)Bn * 112), 256, 0, stream>>>(item_e, item_he, genre_e, gf, is_in, Bn);
  gemm(is_in, is_W1, is_b1, s256, Bn, 256, 112, 1);
  gemm(s256, is_W2, is_b2, is_, Bn, 64, 256, 1);
  k_fusetop<<<egrid((long)Bn * 64), 256, 0, stream>>>(us, is_, xtop, Bn);

  // 6) top MLP
  gemm(xtop, tm_W1, tm_b1, s256, Bn, 256, 388, 1);
  gemm(s256, tm_W2, tm_b2, s128, Bn, 128, 256, 1);
  gemm(s128, tm_W3, tm_b3, t3, Bn, 64, 128, 1);
  k_final<<<egrid(Bn), 256, 0, stream>>>(t3, tm_W4, tm_b4, (float*)d_out, Bn);
}
